// VNGNN_44100724195822
// MI455X (gfx1250) — matrix operand order repaired and checked by reading, not compiled, not run on hardware
//
#include <hip/hip_runtime.h>
#include <hip/hip_bf16.h>

#define N_HID 256
#define N_OUT 40
#define N_OUTP 48   // padded OUT for 16-wide WMMA tiles
#define BN_EPS 1e-5f
#define MT 8        // m-tiles per wave in the GEMM

typedef __attribute__((ext_vector_type(16))) __bf16 v16bf;
typedef __attribute__((ext_vector_type(8)))  float  v8f;

// ---------------------------------------------------------------- utilities
__global__ void k_zero_f32(float* p, int n) {
    int i = blockIdx.x * blockDim.x + threadIdx.x;
    if (i < n) p[i] = 0.0f;
}

__global__ void k_copy_f32(const float* __restrict__ src, float* __restrict__ dst, int n) {
    int i = blockIdx.x * blockDim.x + threadIdx.x;
    if (i < n) dst[i] = src[i];
}

__global__ void k_deg_accum(const int* __restrict__ dst, float* deg, int E) {
    int e = blockIdx.x * blockDim.x + threadIdx.x;
    if (e < E) atomicAdd(&deg[dst[e]], 1.0f);
}

__global__ void k_deg_to_dis(float* d, int n) {
    int i = blockIdx.x * blockDim.x + threadIdx.x;
    if (i < n) d[i] = rsqrtf(d[i] + 1.0f);
}

// ---------------------------------------------------------------- W packing
// Pack W [256 x cols] (f32, row-major) into WMMA B-fragment order (bf16):
//   packed[((ntile*8 + kblk)*2 + hi)*256 + lo*16 + j] = W[(kblk*32+hi*16+j)*cols + ntile*16+lo]
__global__ void k_pack_w(const float* __restrict__ W, __bf16* __restrict__ out,
                         int cols, int ncolPad) {
    int p = blockIdx.x * blockDim.x + threadIdx.x;
    if (p >= N_HID * ncolPad) return;
    int j  =  p        & 15;
    int lo = (p >> 4)  & 15;
    int hi = (p >> 8)  & 1;
    int kb = (p >> 9)  & 7;
    int nt =  p >> 12;
    int k   = kb * 32 + hi * 16 + j;
    int col = nt * 16 + lo;
    out[p] = (col < cols) ? (__bf16)W[k * cols + col] : (__bf16)0.0f;
}

// hbf[i] = bf16(h[i] + vx[i % 256])
__global__ void k_add_vx_convert(const float* __restrict__ h, const float* __restrict__ vx,
                                 __bf16* __restrict__ out, int n) {
    int i = blockIdx.x * blockDim.x + threadIdx.x;
    if (i < n) out[i] = (__bf16)(h[i] + vx[i & 255]);
}

// ---------------------------------------------------------------- WMMA GEMM
// C[nrows x NCOL] (f32) = A[nrows x 256] (bf16) @ W (packed bf16, B-fragment order).
// NCOL compile-time: row strides fold into immediate store offsets.
// grid = (ceil(mtiles/(4*MT)), NCOL/16); block = 128 (4 waves).
template<int NCOL>
__global__ void __launch_bounds__(128)
k_gemm_wmma_bf16(const __bf16* __restrict__ A, const __bf16* __restrict__ Bp,
                 float* __restrict__ C, int nrows) {
    __shared__ __bf16 sB[8 * 2 * 256];                 // 4096 bf16 = 8 KB
    const int ntile = blockIdx.y;

    // cooperative 8KB copy: 512 uint4, 128 threads x 4
    const uint4* __restrict__ gsrc = (const uint4*)(Bp + (size_t)ntile * 4096);
    uint4* ldst = (uint4*)sB;
    #pragma unroll
    for (int i = 0; i < 4; ++i) ldst[threadIdx.x + 128 * i] = gsrc[threadIdx.x + 128 * i];
    __syncthreads();

    const int lane = threadIdx.x & 31;
    const int wave = threadIdx.x >> 5;
    const int hi = lane >> 4, lo = lane & 15;
    const int mtiles = nrows >> 4;

    #pragma unroll 1
    for (int t = 0; t < MT; ++t) {
        int mt = (blockIdx.x * 4 + wave) * MT + t;     // wave-uniform
        if (mt >= mtiles) break;                       // uniform exit: EXEC stays all-1s
        const __bf16* __restrict__ arow = A + (size_t)(mt * 16 + lo) * N_HID;
        v8f acc = {};
        #pragma unroll
        for (int kblk = 0; kblk < 8; ++kblk) {
            // A fragment: elem i -> K=kblk*32+hi*16+i
            union { uint4 u[2]; v16bf v; } ua;
            ua.u[0] = *(const uint4*)(arow + kblk * 32 + hi * 16);
            ua.u[1] = *(const uint4*)(arow + kblk * 32 + hi * 16 + 8);
            // B fragment from LDS: elem j -> K=kblk*32+hi*16+j, col=n0+lo
            union { uint4 u[2]; v16bf v; } ub;
            const uint4* bp = (const uint4*)(sB + ((kblk * 2 + hi) * 256 + lo * 16));
            ub.u[0] = bp[0];
            ub.u[1] = bp[1];
            acc = __builtin_amdgcn_wmma_f32_16x16x32_bf16(
                      false, ua.v, false, ub.v, (short)0, acc, false, false);
        }
        // C/D layout: elem r -> row mt*16 + hi*8 + r, col ntile*16 + lo
        float* crow = C + (size_t)(mt * 16 + hi * 8) * NCOL + ntile * 16 + lo;
        #pragma unroll
        for (int r = 0; r < 8; ++r) crow[r * NCOL] = acc[r];   // imm-offset stores
    }
}

// ---------------------------------------------------------------- aggregation
// agg[r,c] = t[r,c] * dis[r]^2 + bias[c]   (vectorized, float4)
__global__ void k_agg_init(const float* __restrict__ t, int tstride,
                           const float* __restrict__ dis, const float* __restrict__ bias,
                           float* __restrict__ agg, int nrows, int C) {
    int i = blockIdx.x * blockDim.x + threadIdx.x;   // float4 index
    int C4 = C >> 2;
    if (i >= nrows * C4) return;
    int r = i / C4, c = (i - r * C4) << 2;
    float d = dis[r];  float d2 = d * d;
    const float4 v = *(const float4*)(t + (size_t)r * tstride + c);
    const float4 bb = *(const float4*)(bias + c);
    float4 o;
    o.x = v.x * d2 + bb.x; o.y = v.y * d2 + bb.y;
    o.z = v.z * d2 + bb.z; o.w = v.w * d2 + bb.w;
    *(float4*)(agg + (size_t)r * C + c) = o;
}

// per edge: agg[dst,:] += t[src,:] * dis[src]*dis[dst]
// C=256: 64 threads/edge x 4 channels, 4 edges per 256-thread block.
__global__ void k_edge_scatter256(const int* __restrict__ src, const int* __restrict__ dst,
                                  const float* __restrict__ dis, const float* __restrict__ t,
                                  float* __restrict__ agg, int E) {
    int e = blockIdx.x * 4 + (threadIdx.x >> 6);
    if (e >= E) return;
    int c = (threadIdx.x & 63) << 2;
    int s = src[e], d = dst[e];
    float w = dis[s] * dis[d];
    const float4 v = *(const float4*)(t + (size_t)s * N_HID + c);
    float* p = agg + (size_t)d * N_HID + c;
    atomicAdd(p + 0, v.x * w); atomicAdd(p + 1, v.y * w);
    atomicAdd(p + 2, v.z * w); atomicAdd(p + 3, v.w * w);
}

// C=40 (t stride 48): 16 threads/edge (10 active), 16 edges per 256-thread block.
__global__ void k_edge_scatter40(const int* __restrict__ src, const int* __restrict__ dst,
                                 const float* __restrict__ dis, const float* __restrict__ t,
                                 float* __restrict__ agg, int E) {
    int e = blockIdx.x * 16 + (threadIdx.x >> 4);
    int c = (threadIdx.x & 15) << 2;
    if (e >= E || c >= N_OUT) return;
    int s = src[e], d = dst[e];
    float w = dis[s] * dis[d];
    const float4 v = *(const float4*)(t + (size_t)s * N_OUTP + c);
    float* p = agg + (size_t)d * N_OUT + c;
    atomicAdd(p + 0, v.x * w); atomicAdd(p + 1, v.y * w);
    atomicAdd(p + 2, v.z * w); atomicAdd(p + 3, v.w * w);
}

// ---------------------------------------------------------------- batchnorm
__global__ void k_bn_stats(const float* __restrict__ h, float* __restrict__ sums, int nrows) {
    int c = threadIdx.x;
    float s = 0.f, s2 = 0.f;
    for (int r = blockIdx.x; r < nrows; r += gridDim.x) {
        float v = h[(size_t)r * N_HID + c];
        s += v; s2 += v * v;
    }
    atomicAdd(&sums[c], s);
    atomicAdd(&sums[N_HID + c], s2);
}

__global__ void k_bn_apply_relu_pool(float* __restrict__ h, const float* __restrict__ sums,
                                     const float* __restrict__ g, const float* __restrict__ b,
                                     float* __restrict__ pool, int nrows, int rowsPerBlk) {
    int c = threadIdx.x;
    float inv_n = 1.0f / (float)nrows;
    float mu  = sums[c] * inv_n;
    float var = sums[N_HID + c] * inv_n - mu * mu;
    float sc  = rsqrtf(var + BN_EPS) * g[c];
    float sh  = b[c] - mu * sc;
    float accp = 0.f;
    int r0 = blockIdx.x * rowsPerBlk;
    int r1 = min(r0 + rowsPerBlk, nrows);
    for (int r = r0; r < r1; ++r) {
        float v = fmaxf(h[(size_t)r * N_HID + c] * sc + sh, 0.0f);
        h[(size_t)r * N_HID + c] = v;
        accp += v;
    }
    atomicAdd(&pool[c], accp);
}

// ---------------------------------------------------------------- virtual node
__global__ void k_vn_update(const float* __restrict__ pool, const float* __restrict__ vx_in,
                            const float* __restrict__ W, const float* __restrict__ bias,
                            const float* __restrict__ lng, const float* __restrict__ lnb,
                            float* __restrict__ vx_out) {
    __shared__ float in[N_HID];
    __shared__ float red[N_HID];
    int j = threadIdx.x;
    in[j] = pool[j] + vx_in[j];
    __syncthreads();
    float acc = bias[j];
    #pragma unroll 4
    for (int i = 0; i < N_HID; ++i) acc += in[i] * W[i * N_HID + j];
    red[j] = acc; __syncthreads();
    for (int s = N_HID >> 1; s > 0; s >>= 1) {
        if (j < s) red[j] += red[j + s];
        __syncthreads();
    }
    float mu = red[0] * (1.0f / N_HID);
    __syncthreads();
    float d = acc - mu;
    red[j] = d * d; __syncthreads();
    for (int s = N_HID >> 1; s > 0; s >>= 1) {
        if (j < s) red[j] += red[j + s];
        __syncthreads();
    }
    float var = red[0] * (1.0f / N_HID);
    vx_out[j] = fmaxf(d * rsqrtf(var + BN_EPS) * lng[j] + lnb[j], 0.0f);
}

// ---------------------------------------------------------------- log_softmax (40)
__global__ void k_log_softmax40(float* __restrict__ out, int nrows) {
    int row  = blockIdx.x * (blockDim.x >> 5) + (threadIdx.x >> 5);
    int lane = threadIdx.x & 31;
    if (row >= nrows) return;
    float* p = out + (size_t)row * N_OUT;
    float v1 = p[lane];
    bool  has2 = (lane + 32) < N_OUT;
    float v2 = has2 ? p[lane + 32] : -INFINITY;
    float m = fmaxf(v1, v2);
    #pragma unroll
    for (int i = 16; i > 0; i >>= 1) m = fmaxf(m, __shfl_xor(m, i, 32));
    float s = __expf(v1 - m) + (has2 ? __expf(v2 - m) : 0.0f);
    #pragma unroll
    for (int i = 16; i > 0; i >>= 1) s += __shfl_xor(s, i, 32);
    float lse = m + __logf(s);
    p[lane] = v1 - lse;
    if (has2) p[lane + 32] = v2 - lse;
}

// ================================================================ launch
extern "C" void kernel_launch(void* const* d_in, const int* in_sizes, int n_in,
                              void* d_out, int out_size, void* d_ws, size_t ws_size,
                              hipStream_t stream) {
    const float* x    = (const float*)d_in[0];
    const int*   ei   = (const int*)d_in[1];
    const float* W0   = (const float*)d_in[2];
    const float* b0   = (const float*)d_in[3];
    const float* W1   = (const float*)d_in[4];
    const float* b1   = (const float*)d_in[5];
    const float* W2   = (const float*)d_in[6];
    const float* b2   = (const float*)d_in[7];
    const float* bng0 = (const float*)d_in[8];
    const float* bnb0 = (const float*)d_in[9];
    const float* bng1 = (const float*)d_in[10];
    const float* bnb1 = (const float*)d_in[11];
    const float* vne  = (const float*)d_in[12];
    const float* vW0  = (const float*)d_in[13];
    const float* vb0  = (const float*)d_in[14];
    const float* lg0  = (const float*)d_in[15];
    const float* lb0  = (const float*)d_in[16];
    const float* vW1  = (const float*)d_in[17];
    const float* vb1  = (const float*)d_in[18];
    const float* lg1  = (const float*)d_in[19];
    const float* lb1  = (const float*)d_in[20];

    const int N = in_sizes[0] / N_HID;
    const int E = in_sizes[1] / 2;
    const int* esrc = ei;
    const int* edst = ei + E;

    // ---- workspace carve-out
    char* ws = (char*)d_ws;
    size_t off = 0;
    auto carve = [&](size_t bytes) -> char* {
        char* p = ws + off;
        off = (off + bytes + 255) & ~(size_t)255;
        return p;
    };
    float*  dis  = (float*)carve((size_t)N * 4);
    __bf16* hbf  = (__bf16*)carve((size_t)N * N_HID * 2);
    float*  bufT = (float*)carve((size_t)N * N_HID * 4);
    float*  bufH = (float*)carve((size_t)N * N_HID * 4);
    __bf16* wb0  = (__bf16*)carve((size_t)N_HID * N_HID * 2);
    __bf16* wb1  = (__bf16*)carve((size_t)N_HID * N_HID * 2);
    __bf16* wb2  = (__bf16*)carve((size_t)N_HID * N_OUTP * 2);
    float*  sums = (float*)carve(2 * N_HID * 4);
    float*  pool = (float*)carve(N_HID * 4);
    float*  vxA  = (float*)carve(N_HID * 4);
    float*  vxB  = (float*)carve(N_HID * 4);
    (void)ws_size; (void)n_in; (void)out_size;

    float* logits = (float*)d_out;   // [N, 40]

    const int TPB = 256;
    dim3 blk(TPB);

    // ---- degree / symmetric normalization
    k_zero_f32<<<dim3((N + TPB - 1) / TPB), blk, 0, stream>>>(dis, N);
    k_deg_accum<<<dim3((E + TPB - 1) / TPB), blk, 0, stream>>>(edst, dis, E);
    k_deg_to_dis<<<dim3((N + TPB - 1) / TPB), blk, 0, stream>>>(dis, N);

    // ---- weights -> packed bf16 B-fragment layout (once)
    k_pack_w<<<dim3((N_HID * N_HID + TPB - 1) / TPB), blk, 0, stream>>>(W0, wb0, N_HID, N_HID);
    k_pack_w<<<dim3((N_HID * N_HID + TPB - 1) / TPB), blk, 0, stream>>>(W1, wb1, N_HID, N_HID);
    k_pack_w<<<dim3((N_HID * N_OUTP + TPB - 1) / TPB), blk, 0, stream>>>(W2, wb2, N_OUT, N_OUTP);

    // ---- vx <- vn_emb
    k_copy_f32<<<dim3(1), blk, 0, stream>>>(vne, vxA, N_HID);

    const int nElems   = N * N_HID;
    const int gElems   = (nElems + TPB - 1) / TPB;
    const int mtiles   = N >> 4;
    const int gGemmX   = (mtiles + 4 * MT - 1) / (4 * MT);
    const dim3 gGemm256(gGemmX, N_HID >> 4);
    const dim3 gGemm48 (gGemmX, N_OUTP >> 4);
    const int gAgg256  = (N * (N_HID >> 2) + TPB - 1) / TPB;
    const int gScat256 = (E + 3) / 4;
    const int gScat40  = (E + 15) / 16;
    const int gBnApply = (N + 255) / 256;

    // ================= layer 0 =================
    k_add_vx_convert<<<dim3(gElems), blk, 0, stream>>>(x, vxA, hbf, nElems);
    k_gemm_wmma_bf16<N_HID><<<gGemm256, dim3(128), 0, stream>>>(hbf, wb0, bufT, N);
    k_agg_init<<<dim3(gAgg256), blk, 0, stream>>>(bufT, N_HID, dis, b0, bufH, N, N_HID);
    k_edge_scatter256<<<dim3(gScat256), blk, 0, stream>>>(esrc, edst, dis, bufT, bufH, E);
    k_zero_f32<<<dim3(2), blk, 0, stream>>>(sums, 2 * N_HID);
    k_zero_f32<<<dim3(1), blk, 0, stream>>>(pool, N_HID);
    k_bn_stats<<<dim3(512), blk, 0, stream>>>(bufH, sums, N);
    k_bn_apply_relu_pool<<<dim3(gBnApply), blk, 0, stream>>>(bufH, sums, bng0, bnb0, pool, N, 256);
    k_vn_update<<<dim3(1), blk, 0, stream>>>(pool, vxA, vW0, vb0, lg0, lb0, vxB);

    // ================= layer 1 =================
    k_add_vx_convert<<<dim3(gElems), blk, 0, stream>>>(bufH, vxB, hbf, nElems);
    k_gemm_wmma_bf16<N_HID><<<gGemm256, dim3(128), 0, stream>>>(hbf, wb1, bufT, N);
    k_agg_init<<<dim3(gAgg256), blk, 0, stream>>>(bufT, N_HID, dis, b1, bufH, N, N_HID);
    k_edge_scatter256<<<dim3(gScat256), blk, 0, stream>>>(esrc, edst, dis, bufT, bufH, E);
    k_zero_f32<<<dim3(2), blk, 0, stream>>>(sums, 2 * N_HID);
    k_zero_f32<<<dim3(1), blk, 0, stream>>>(pool, N_HID);
    k_bn_stats<<<dim3(512), blk, 0, stream>>>(bufH, sums, N);
    k_bn_apply_relu_pool<<<dim3(gBnApply), blk, 0, stream>>>(bufH, sums, bng1, bnb1, pool, N, 256);
    k_vn_update<<<dim3(1), blk, 0, stream>>>(pool, vxB, vW1, vb1, lg1, lb1, vxA);

    // ================= layer 2 (output) =================
    k_add_vx_convert<<<dim3(gElems), blk, 0, stream>>>(bufH, vxA, hbf, nElems);
    k_gemm_wmma_bf16<N_OUTP><<<gGemm48, dim3(128), 0, stream>>>(hbf, wb2, bufT, N);
    {
        int gAgg40 = (N * (N_OUT >> 2) + TPB - 1) / TPB;
        k_agg_init<<<dim3(gAgg40), blk, 0, stream>>>(bufT, N_OUTP, dis, b2, logits, N, N_OUT);
        k_edge_scatter40<<<dim3(gScat40), blk, 0, stream>>>(esrc, edst, dis, bufT, logits, E);
    }
    k_log_softmax40<<<dim3((N + 7) / 8), blk, 0, stream>>>(logits, N);
}
